// MultiHeadAttention_84404697301545
// MI455X (gfx1250) — matrix operand order repaired and checked by reading, not compiled, not run on hardware
//
#include <hip/hip_runtime.h>

// ---------------------------------------------------------------------------
// MHA forward (causal + RoPE) for gfx1250 using bf16 WMMA (16x16x32, f32 acc),
// with double-buffered async K/V staging into LDS (ASYNCcnt path).
//   DIM=1024, HEADS=16, HD=64, SEQ=2048, BATCH=4
// Workspace layout (bf16): Xb | Wq | Wk | Wv | Wo | Q | K | Vt | AO  (~88 MB)
// ---------------------------------------------------------------------------

#define NDIM   1024
#define NHEADS 16
#define HD     64
#define NSEQ   2048
#define NB     4
#define NROWS  (NB * NSEQ)   // 8192

typedef __attribute__((ext_vector_type(16))) __bf16 v16bf;
typedef __attribute__((ext_vector_type(8)))  __bf16 v8bf;
typedef __attribute__((ext_vector_type(8)))  float  v8f;

static __device__ __forceinline__ __bf16 f2bf(float f) {
  union { float f; unsigned u; } v; v.f = f;
  unsigned r = v.u + 0x7FFFu + ((v.u >> 16) & 1u);      // round-to-nearest-even
  unsigned short h = (unsigned short)(r >> 16);
  return __builtin_bit_cast(__bf16, h);
}

static __device__ __forceinline__ v8f wmma_bf16(v16bf a, v16bf b, v8f c) {
  return __builtin_amdgcn_wmma_f32_16x16x32_bf16(false, a, false, b,
                                                 (short)0, c, false, false);
}

// Pack two contiguous 8-element chunks into a wave32 A operand in contiguous k order (B loads match):
// element j of v16bf holds K = j + 16*half  (half = lane>>4)
static __device__ __forceinline__ v16bf load_a(const __bf16* rowp, int hf) {
  v8bf lo = *(const v8bf*)(rowp + 16 * hf);
  v8bf hi = *(const v8bf*)(rowp + 16 * hf + 8);
  v16bf a;
#pragma unroll
  for (int j = 0; j < 8; ++j) { a[j] = lo[j]; a[j + 8] = hi[j]; }
  return a;
}

// Async global->LDS 16B copy, tracked by ASYNCcnt (CDNA5 ISA).
// lds_off: byte offset within the workgroup's LDS allocation (low 32 bits of the
// generic pointer to a __shared__ object per the flat-address aperture mapping).
static __device__ __forceinline__ void async_g2l_b128(unsigned lds_off,
                                                      const void* gptr) {
  unsigned long long ga = (unsigned long long)(uintptr_t)gptr;
  asm volatile("global_load_async_to_lds_b128 %0, %1, off"
               :: "v"(lds_off), "v"(ga)
               : "memory");
}

static __device__ __forceinline__ void wait_asynccnt0() {
  asm volatile("s_wait_asynccnt 0x0" ::: "memory");
}

// --------------------------- cast f32 -> bf16 ------------------------------
__global__ __launch_bounds__(256) void cast_bf16_kernel(
    const float* __restrict__ src, __bf16* __restrict__ dst, int n) {
  int i = blockIdx.x * 256 + threadIdx.x;
  if (i < n) dst[i] = f2bf(src[i]);
}

// --------------------- QKV projection + RoPE + layout ----------------------
// blockIdx.y: 0=Q, 1=K, 2=V.  Each wave computes a 16x64 tile of x@W^T.
__global__ __launch_bounds__(256) void qkv_kernel(
    const __bf16* __restrict__ X,                     // [NROWS][NDIM]
    const __bf16* __restrict__ Wq, const __bf16* __restrict__ Wk,
    const __bf16* __restrict__ Wv,
    const float* __restrict__ cosT, const float* __restrict__ sinT, // [NSEQ][HD]
    __bf16* __restrict__ Qo,                          // [B*H][NSEQ][HD]
    __bf16* __restrict__ Ko,                          // [B*H][NSEQ][HD]
    __bf16* __restrict__ Vt) {                        // [B*H][HD][NSEQ]
  const int which = blockIdx.y;
  const __bf16* W = (which == 0) ? Wq : (which == 1) ? Wk : Wv;
  const int wave = threadIdx.x >> 5;
  const int lane = threadIdx.x & 31;
  const int hf   = lane >> 4;
  const int l16  = lane & 15;
  const int tile = blockIdx.x * 8 + wave;   // 8192 tiles
  const int mt = tile >> 4;                 // 512 row tiles
  const int nt = tile & 15;                 // 16 col tiles (== head index)
  const int m0 = mt * 16;
  const int n0 = nt * 64;

  v8f acc[4] = {};
  const __bf16* Arow = X + (size_t)(m0 + l16) * NDIM;
  for (int k0 = 0; k0 < NDIM; k0 += 32) {
    v16bf a = load_a(Arow + k0, hf);
#pragma unroll
    for (int i = 0; i < 4; ++i) {
      // B[k][n] = W[n][k]: lane l16 holds column n, 16 consecutive k -> 32B load
      const v16bf* bp =
          (const v16bf*)(W + (size_t)(n0 + i * 16 + l16) * NDIM + k0 + 16 * hf);
      acc[i] = wmma_bf16(a, *bp, acc[i]);
    }
  }

  const int srow = m0 % NSEQ;   // tiles never cross batch (2048 % 16 == 0)
  const int b    = m0 / NSEQ;
  const int h    = nt;

  if (which < 2) {              // RoPE: q*cos + rotate_half(q)*sin
    v8f p0 = acc[0], p1 = acc[1], p2 = acc[2], p3 = acc[3];
#pragma unroll
    for (int i = 0; i < 4; ++i) {
      int d = i * 16 + l16;
#pragma unroll
      for (int r = 0; r < 8; ++r) {
        int s = srow + r + 8 * hf;
        float c  = cosT[s * HD + d];
        float sn = sinT[s * HD + d];
        float rot = (i == 0) ? -p2[r] : (i == 1) ? -p3[r] : (i == 2) ? p0[r] : p1[r];
        acc[i][r] = acc[i][r] * c + rot * sn;
      }
    }
    __bf16* dst = ((which == 0) ? Qo : Ko) + (size_t)(b * NHEADS + h) * NSEQ * HD;
#pragma unroll
    for (int i = 0; i < 4; ++i)
#pragma unroll
      for (int r = 0; r < 8; ++r)
        dst[(size_t)(srow + r + 8 * hf) * HD + i * 16 + l16] = f2bf(acc[i][r]);
  } else {                      // V stored transposed: Vt[bh][d][s]
    __bf16* dst = Vt + (size_t)(b * NHEADS + h) * HD * NSEQ;
#pragma unroll
    for (int i = 0; i < 4; ++i)
#pragma unroll
      for (int r = 0; r < 8; ++r)
        dst[(size_t)(i * 16 + l16) * NSEQ + srow + r + 8 * hf] = f2bf(acc[i][r]);
  }
}

// -------------------- flash attention (causal, online softmax) -------------
// grid (NSEQ/128, B*H); each wave owns 16 query rows; the whole workgroup
// shares double-buffered 64-key K/V tiles staged in LDS by async DMA.
// The kb loop is block-uniform (waves past their causal bound contribute
// exp(-inf)=0 rows), which makes the __syncthreads() legal.
__global__ __launch_bounds__(256) void attn_kernel(
    const __bf16* __restrict__ Q, const __bf16* __restrict__ K,
    const __bf16* __restrict__ Vt, __bf16* __restrict__ AO) {
  __shared__ alignas(128) __bf16 Kbuf[2][64 * 64];   // 16 KB  [key][d]
  __shared__ alignas(128) __bf16 Vbuf[2][64 * 64];   // 16 KB  [d][key]
  __shared__ alignas(64)  __bf16 Pbuf[8][16 * 64];   // 16 KB per-wave P staging
  const int tid  = threadIdx.x;
  const int wave = tid >> 5;
  const int lane = tid & 31;
  const int hf   = lane >> 4;
  const int l16  = lane & 15;
  const int bh   = blockIdx.y;
  const int q0   = blockIdx.x * 128 + wave * 16;
  const int q_hi = blockIdx.x * 128 + 127;           // block-uniform causal bound

  const __bf16* Qh = Q  + (size_t)bh * NSEQ * HD;
  const __bf16* Kh = K  + (size_t)bh * NSEQ * HD;
  const __bf16* Vh = Vt + (size_t)bh * HD * NSEQ;

  // Q tile (16x64) in A layout, loaded once
  v16bf qa[2];
  const __bf16* Qrow = Qh + (size_t)(q0 + l16) * HD;
#pragma unroll
  for (int kk = 0; kk < 2; ++kk) qa[kk] = load_a(Qrow + kk * 32, hf);

  v8f o[4] = {};
  float mrow[8], lrow[8];
#pragma unroll
  for (int r = 0; r < 8; ++r) { mrow[r] = -3.0e38f; lrow[r] = 0.0f; }
  const float scale = 0.125f;   // 1/sqrt(64)

  const unsigned koff0 = (unsigned)(uintptr_t)&Kbuf[0][0];
  const unsigned voff0 = (unsigned)(uintptr_t)&Vbuf[0][0];
  const int ldrow = tid >> 3;          // 0..31
  const int ldch  = (tid & 7) * 8;     // element chunk within a 64-elem row

  // Cooperative async stage of the 64x64 K tile (row-major [key][d]) and the
  // 64x64 V tile (transposed, [d][key]) for key block kb into buffer `buf`.
  auto stage = [&](int buf, int kb) {
    unsigned kbase = koff0 + (unsigned)buf * 64 * 64 * 2;
    unsigned vbase = voff0 + (unsigned)buf * 64 * 64 * 2;
#pragma unroll
    for (int pass = 0; pass < 2; ++pass) {
      int row = ldrow + pass * 32;
      async_g2l_b128(kbase + (unsigned)(row * 64 + ldch) * 2,
                     Kh + (size_t)(kb + row) * HD + ldch);
      async_g2l_b128(vbase + (unsigned)(row * 64 + ldch) * 2,
                     Vh + (size_t)row * NSEQ + kb + ldch);
    }
  };

  stage(0, 0);
  wait_asynccnt0();
  __syncthreads();

  int cur = 0;
  for (int kb = 0; kb <= q_hi; kb += 64) {
    if (kb + 64 <= q_hi) stage(cur ^ 1, kb + 64);   // prefetch next tile

    const __bf16* kt = &Kbuf[cur][0];
    const __bf16* vt = &Vbuf[cur][0];

    // S = Q @ K^T over this 64-key block (B operand from LDS)
    v8f sacc[4] = {};
#pragma unroll
    for (int i = 0; i < 4; ++i)
#pragma unroll
      for (int kk = 0; kk < 2; ++kk) {
        const v16bf* bp =
            (const v16bf*)(kt + (i * 16 + l16) * 64 + kk * 32 + 16 * hf);
        sacc[i] = wmma_bf16(qa[kk], *bp, sacc[i]);
      }
    // scale + causal mask + online softmax update
#pragma unroll
    for (int r = 0; r < 8; ++r) {
      const int qi = q0 + r + 8 * hf;
      float mx = -3.0e38f;
#pragma unroll
      for (int i = 0; i < 4; ++i) {
        float s = sacc[i][r] * scale;
        if (kb + i * 16 + l16 > qi) s = -3.0e38f;
        sacc[i][r] = s;
        mx = fmaxf(mx, s);
      }
#pragma unroll
      for (int m = 1; m < 16; m <<= 1) mx = fmaxf(mx, __shfl_xor(mx, m, 32));
      float mnew = fmaxf(mrow[r], mx);
      float corr = __expf(mrow[r] - mnew);
      float psum = 0.0f;
#pragma unroll
      for (int i = 0; i < 4; ++i) {
        float p = __expf(sacc[i][r] - mnew);
        sacc[i][r] = p;
        psum += p;
      }
#pragma unroll
      for (int m = 1; m < 16; m <<= 1) psum += __shfl_xor(psum, m, 32);
      lrow[r] = lrow[r] * corr + psum;
      mrow[r] = mnew;
#pragma unroll
      for (int i = 0; i < 4; ++i) o[i][r] *= corr;
    }
    // Stage P (C layout) into wave-private LDS as 16x64 row-major bf16
    __bf16* pbuf = Pbuf[wave];
#pragma unroll
    for (int i = 0; i < 4; ++i)
#pragma unroll
      for (int r = 0; r < 8; ++r)
        pbuf[(r + 8 * hf) * 64 + i * 16 + l16] = f2bf(sacc[i][r]);
    // O += P @ V   (B operand from the LDS V tile: contiguous per lane)
#pragma unroll
    for (int kk = 0; kk < 2; ++kk) {
      v16bf pa = load_a(pbuf + l16 * 64 + kk * 32, hf);
#pragma unroll
      for (int i = 0; i < 4; ++i) {
        const v16bf* vp =
            (const v16bf*)(vt + (i * 16 + l16) * 64 + kk * 32 + 16 * hf);
        o[i] = wmma_bf16(pa, *vp, o[i]);
      }
    }

    wait_asynccnt0();   // next tile fully landed in LDS
    __syncthreads();    // everyone done reading cur + done staging
    cur ^= 1;
  }

  // normalize and write attn output as [b][s][h*64+d] bf16
  const int b = bh / NHEADS, h = bh % NHEADS;
#pragma unroll
  for (int i = 0; i < 4; ++i)
#pragma unroll
    for (int r = 0; r < 8; ++r) {
      int s = q0 + r + 8 * hf;
      int d = i * 16 + l16;
      AO[((size_t)b * NSEQ + s) * NDIM + h * HD + d] = f2bf(o[i][r] / lrow[r]);
    }
}

// --------------------------- output projection -----------------------------
__global__ __launch_bounds__(256) void out_proj_kernel(
    const __bf16* __restrict__ AO, const __bf16* __restrict__ Wo,
    float* __restrict__ out) {
  const int wave = threadIdx.x >> 5;
  const int lane = threadIdx.x & 31;
  const int hf = lane >> 4, l16 = lane & 15;
  const int tile = blockIdx.x * 8 + wave;
  const int m0 = (tile >> 4) * 16;
  const int n0 = (tile & 15) * 64;
  v8f acc[4] = {};
  const __bf16* Arow = AO + (size_t)(m0 + l16) * NDIM;
  for (int k0 = 0; k0 < NDIM; k0 += 32) {
    v16bf a = load_a(Arow + k0, hf);
#pragma unroll
    for (int i = 0; i < 4; ++i) {
      const v16bf* bp =
          (const v16bf*)(Wo + (size_t)(n0 + i * 16 + l16) * NDIM + k0 + 16 * hf);
      acc[i] = wmma_bf16(a, *bp, acc[i]);
    }
  }
#pragma unroll
  for (int i = 0; i < 4; ++i)
#pragma unroll
    for (int r = 0; r < 8; ++r)
      out[(size_t)(m0 + r + 8 * hf) * NDIM + n0 + i * 16 + l16] = acc[i][r];
}

// ---------------------------------------------------------------------------
extern "C" void kernel_launch(void* const* d_in, const int* in_sizes, int n_in,
                              void* d_out, int out_size, void* d_ws, size_t ws_size,
                              hipStream_t stream) {
  const float* x    = (const float*)d_in[0];
  const float* Wq   = (const float*)d_in[1];
  const float* Wk   = (const float*)d_in[2];
  const float* Wv   = (const float*)d_in[3];
  const float* Wo   = (const float*)d_in[4];
  const float* cosT = (const float*)d_in[5];
  const float* sinT = (const float*)d_in[6];
  // d_in[7] (mask) unused: causal mask is applied analytically.

  const size_t RD = (size_t)NROWS * NDIM;       // 8,388,608
  const size_t WW = (size_t)NDIM * NDIM;        // 1,048,576
  const size_t QS = (size_t)NB * NHEADS * NSEQ * HD;  // == RD

  __bf16* p   = (__bf16*)d_ws;                  // ~88 MB of bf16 scratch
  __bf16* Xb  = p;  p += RD;
  __bf16* Wqb = p;  p += WW;
  __bf16* Wkb = p;  p += WW;
  __bf16* Wvb = p;  p += WW;
  __bf16* Wob = p;  p += WW;
  __bf16* Qb  = p;  p += QS;
  __bf16* Kb  = p;  p += QS;
  __bf16* Vtb = p;  p += QS;
  __bf16* AOb = p;  p += RD;

  cast_bf16_kernel<<<(int)((RD + 255) / 256), 256, 0, stream>>>(x, Xb, (int)RD);
  cast_bf16_kernel<<<(int)((WW + 255) / 256), 256, 0, stream>>>(Wq, Wqb, (int)WW);
  cast_bf16_kernel<<<(int)((WW + 255) / 256), 256, 0, stream>>>(Wk, Wkb, (int)WW);
  cast_bf16_kernel<<<(int)((WW + 255) / 256), 256, 0, stream>>>(Wv, Wvb, (int)WW);
  cast_bf16_kernel<<<(int)((WW + 255) / 256), 256, 0, stream>>>(Wo, Wob, (int)WW);

  qkv_kernel<<<dim3(NROWS / 16 * (NDIM / 64) / 8, 3), 256, 0, stream>>>(
      Xb, Wqb, Wkb, Wvb, cosT, sinT, Qb, Kb, Vtb);

  attn_kernel<<<dim3(NSEQ / 128, NB * NHEADS), 256, 0, stream>>>(Qb, Kb, Vtb, AOb);

  out_proj_kernel<<<NROWS / 16 * (NDIM / 64) / 8, 256, 0, stream>>>(
      AOb, Wob, (float*)d_out);
}
